// DyHGCN_87789131530592
// MI455X (gfx1250) — compile-verified, matrix operand order checked by reading, not run on hardware
//
#include <hip/hip_runtime.h>
#include <hip/hip_bf16.h>

typedef __attribute__((ext_vector_type(8)))  float  v8f;
typedef __attribute__((ext_vector_type(16))) __bf16 v16bf;
typedef __attribute__((ext_vector_type(8)))  __bf16 v8bf;

#define DEV static __device__ __forceinline__

DEV unsigned short f2bf(float f) {
  unsigned int u = __float_as_uint(f);
  return (unsigned short)((u + 0x7FFFu + ((u >> 16) & 1u)) >> 16);
}
DEV float bf2f(unsigned int lo16) { return __uint_as_float(lo16 << 16); }
DEV void atomAddF(float* p, float v) {
  __hip_atomic_fetch_add(p, v, __ATOMIC_RELAXED, __HIP_MEMORY_SCOPE_AGENT);
}

// ---------------- elementwise helpers ----------------
__global__ void k_f32_to_bf16(const float* __restrict__ in,
                              unsigned short* __restrict__ out, int n) {
  int i = blockIdx.x * blockDim.x + threadIdx.x;
  if (i < n) out[i] = f2bf(in[i]);
}

__global__ void k_relu_bf16(const float* __restrict__ in,
                            unsigned short* __restrict__ out, int n) {
  int i = blockIdx.x * blockDim.x + threadIdx.x;
  if (i < n) out[i] = f2bf(fmaxf(in[i], 0.0f));
}

// W [K][Ncols] row-major -> out [Ncols][K] bf16 (WMMA B operand layout source)
__global__ void k_transpose_bf16(const float* __restrict__ W,
                                 unsigned short* __restrict__ out,
                                 int K, int Ncols) {
  int i = blockIdx.x * blockDim.x + threadIdx.x;
  if (i >= K * Ncols) return;
  int n = i / K, k = i - n * K;
  out[i] = f2bf(W[k * Ncols + n]);
}

__global__ void k_bias_sum(const float* __restrict__ a, const float* __restrict__ b,
                           float* __restrict__ out, int n) {
  int i = blockIdx.x * blockDim.x + threadIdx.x;
  if (i < n) out[i] = a[i] + b[i];
}

// ---------------- degree / normalization ----------------
__global__ void k_deg_init(float* deg, int n) {
  int i = blockIdx.x * blockDim.x + threadIdx.x;
  if (i < n) deg[i] = 1.0f;  // self loop
}
__global__ void k_deg_acc(float* __restrict__ deg, const long long* __restrict__ tgt, int E) {
  int stride = gridDim.x * blockDim.x;
  for (int e = blockIdx.x * blockDim.x + threadIdx.x; e < E; e += stride)
    atomAddF(&deg[(int)tgt[e]], 1.0f);
}
__global__ void k_deg_rsqrt(float* deg, int n) {
  int i = blockIdx.x * blockDim.x + threadIdx.x;
  if (i < n) deg[i] = rsqrtf(deg[i]);  // deg >= 1 always
}

// ---------------- WMMA GEMM: C[M,ncols] = A[M,128] x B[128,ncols] ----------------
// A: bf16 row-major [M,128]. Bt: bf16 [ncols,128], Bt[n][k] = B[k][n].
// Per block: 8 waves, wave w -> column tile (blockIdx.y*8 + w), row tile blockIdx.x.
__global__ __launch_bounds__(256) void k_gemm_bf16(
    const unsigned short* __restrict__ A, const unsigned short* __restrict__ Bt,
    const float* __restrict__ bias, float* __restrict__ C, int M, int ncols) {
  int wave = threadIdx.x >> 5;
  int lane = threadIdx.x & 31;
  int mtile = blockIdx.x;
  int ntile = blockIdx.y * 8 + wave;
  if (ntile * 16 >= ncols) return;  // whole-wave exit: EXEC stays all-ones for WMMA

  int mrow = mtile * 16 + (lane & 15);
  int ncol = ntile * 16 + (lane & 15);
  int kb = (lane & 16) ? 8 : 0;  // ISA 16-bit A/B operand K-split across lane halves

  const __bf16* Arow = (const __bf16*)A + (size_t)mrow * 128 + kb;
  const __bf16* Brow = (const __bf16*)Bt + (size_t)ncol * 128 + kb;

  v8f c = {};
#pragma unroll
  for (int kt = 0; kt < 4; ++kt) {
    int k0 = kt * 32;
    v8bf alo = *(const v8bf*)(Arow + k0);
    v8bf ahi = *(const v8bf*)(Arow + k0 + 16);
    v8bf blo = *(const v8bf*)(Brow + k0);
    v8bf bhi = *(const v8bf*)(Brow + k0 + 16);
    v16bf a, b;
#pragma unroll
    for (int i = 0; i < 8; ++i) {
      a[i] = alo[i]; a[8 + i] = ahi[i];
      b[i] = blo[i]; b[8 + i] = bhi[i];
    }
    c = __builtin_amdgcn_wmma_f32_16x16x32_bf16(false, a, false, b,
                                                (short)0, c, false, false);
  }

  int mb = (lane & 16) ? 8 : 0;  // C/D layout: VGPR r -> row r (lanes 0-15) / 8+r (16-31)
  float bn = bias ? bias[ncol] : 0.0f;
#pragma unroll
  for (int r = 0; r < 8; ++r)
    C[(size_t)(mtile * 16 + mb + r) * ncols + ncol] = c[r] + bn;
}

// ---------------- edge scatter (one wave32 per edge) ----------------
__global__ void k_scatter128(const float* __restrict__ h, const long long* __restrict__ src,
                             const long long* __restrict__ tgt, const float* __restrict__ dinv,
                             float* __restrict__ out, int E) {
  int lane = threadIdx.x & 31;
  int gw = (blockIdx.x * blockDim.x + threadIdx.x) >> 5;
  int nw = (gridDim.x * blockDim.x) >> 5;
  for (int e = gw; e < E; e += nw) {
    int s = (int)src[e], t = (int)tgt[e];
    float norm = dinv[s] * dinv[t];
    float4 v = ((const float4*)(h + (size_t)s * 128))[lane];
    float* op = out + (size_t)t * 128 + lane * 4;
    atomAddF(op + 0, norm * v.x);
    atomAddF(op + 1, norm * v.y);
    atomAddF(op + 2, norm * v.z);
    atomAddF(op + 3, norm * v.w);
  }
}

__global__ void k_scatter64(const float* __restrict__ h, const long long* __restrict__ src,
                            const long long* __restrict__ tgt, const float* __restrict__ dinv,
                            float* __restrict__ out, int E) {
  int lane = threadIdx.x & 31;
  int gw = (blockIdx.x * blockDim.x + threadIdx.x) >> 5;
  int nw = (gridDim.x * blockDim.x) >> 5;
  for (int e = gw; e < E; e += nw) {
    int s = (int)src[e], t = (int)tgt[e];
    float norm = dinv[s] * dinv[t];
    float2 v = ((const float2*)(h + (size_t)s * 64))[lane];
    float* op = out + (size_t)t * 64 + lane * 2;
    atomAddF(op + 0, norm * v.x);
    atomAddF(op + 1, norm * v.y);
  }
}

// out_bf16 = bf16(relu(agg + dinv^2 * t + b))   (adds the self-loop term + bias)
__global__ void k_finalize_bf16(const float* __restrict__ agg, const float* __restrict__ t,
                                const float* __restrict__ dinv, const float* __restrict__ b,
                                unsigned short* __restrict__ outb, int total) {
  int i = blockIdx.x * blockDim.x + threadIdx.x;
  if (i >= total) return;
  int row = i >> 7, f = i & 127;
  float dv = dinv[row];
  float v = agg[i] + dv * dv * t[i] + b[f];
  outb[i] = f2bf(fmaxf(v, 0.0f));
}

// acc += relu(agg + dinv^2 * t + b)
__global__ void k_finalize_acc(const float* __restrict__ agg, const float* __restrict__ t,
                               const float* __restrict__ dinv, const float* __restrict__ b,
                               float* __restrict__ acc, int total) {
  int i = blockIdx.x * blockDim.x + threadIdx.x;
  if (i >= total) return;
  int row = i >> 7, f = i & 127;
  float dv = dinv[row];
  float v = agg[i] + dv * dv * t[i] + b[f];
  acc[i] += fmaxf(v, 0.0f);
}

// ---------------- persistent single-WGP LSTM scan ----------------
// G[T,512] precomputed (x@Wih.T + bih + bhh). Whh cached in LDS as packed bf16
// pairs with stride-65-dword rows (consecutive j -> consecutive banks, no conflicts).
__global__ __launch_bounds__(1024) void k_lstm(const float* __restrict__ G,
                                               const float* __restrict__ Whh,
                                               float* __restrict__ hs, int T) {
  __shared__ unsigned int s_whh[512 * 65];  // 133,120 B of the 320 KB WGP LDS
  __shared__ float s_h[128];
  __shared__ float s_part[1024];
  __shared__ float s_act[512];

  int tid = threadIdx.x;
  for (int idx = tid; idx < 512 * 64; idx += 1024) {
    int j = idx >> 6, kp = idx & 63;
    const float* wr = Whh + (size_t)j * 128 + kp * 2;
    unsigned int u = ((unsigned int)f2bf(wr[1]) << 16) | (unsigned int)f2bf(wr[0]);
    s_whh[j * 65 + kp] = u;
  }
  if (tid < 128) s_h[tid] = 0.0f;
  float c = 0.0f;  // cell state lives in lanes tid < 128
  __syncthreads();

  int j = tid & 511;     // gate index
  int half = tid >> 9;   // which 64-wide half of the dot product
  const unsigned int* wrow = s_whh + j * 65 + half * 32;

  for (int t = 0; t < T; ++t) {
    const float* hv = s_h + half * 64;  // same-address broadcast within wave
    float acc = 0.0f;
#pragma unroll
    for (int kk = 0; kk < 32; ++kk) {
      unsigned int u = wrow[kk];
      acc = fmaf(bf2f(u & 0xFFFFu), hv[2 * kk], acc);
      acc = fmaf(bf2f(u >> 16), hv[2 * kk + 1], acc);
    }
    s_part[tid] = acc;
    __syncthreads();
    if (half == 0) {
      float gate = G[(size_t)t * 512 + j] + s_part[j] + s_part[j + 512];
      float a = (j < 256 || j >= 384) ? (1.0f / (1.0f + __expf(-gate)))  // i, f, o
                                      : tanhf(gate);                     // g
      s_act[j] = a;
    }
    __syncthreads();
    if (tid < 128) {
      float iv = s_act[tid], fv = s_act[128 + tid];
      float gv = s_act[256 + tid], ov = s_act[384 + tid];
      c = fv * c + iv * gv;
      float h = ov * tanhf(c);
      s_h[tid] = h;
      hs[(size_t)t * 128 + tid] = h;
    }
    __syncthreads();
  }
}

// ---------------- final GCN + log_softmax (one wave per row) ----------------
__global__ void k_final_logsoftmax(const float* __restrict__ aggo, const float* __restrict__ y0,
                                   const float* __restrict__ dinvc, const float* __restrict__ bo,
                                   float* __restrict__ out, int N) {
  int lane = threadIdx.x & 31;
  int row = (blockIdx.x * blockDim.x + threadIdx.x) >> 5;
  if (row >= N) return;
  float dv = dinvc[row];
  float d2 = dv * dv;
  size_t base = (size_t)row * 64;
  float z0 = aggo[base + lane]      + d2 * y0[base + lane]      + bo[lane];
  float z1 = aggo[base + 32 + lane] + d2 * y0[base + 32 + lane] + bo[32 + lane];
  float m = fmaxf(z0, z1);
  for (int off = 16; off; off >>= 1) m = fmaxf(m, __shfl_xor(m, off));
  float s = __expf(z0 - m) + __expf(z1 - m);
  for (int off = 16; off; off >>= 1) s += __shfl_xor(s, off);
  float lse = m + __logf(s);
  out[base + lane]      = z0 - lse;
  out[base + 32 + lane] = z1 - lse;
}

// ---------------- host orchestration ----------------
extern "C" void kernel_launch(void* const* d_in, const int* in_sizes, int n_in,
                              void* d_out, int out_size, void* d_ws, size_t ws_size,
                              hipStream_t stream) {
  constexpr int N = 20000, E = 640000;
  constexpr int NF = N * 128;

  const float* x = (const float*)d_in[0];
  const long long* ei[3] = {(const long long*)d_in[1], (const long long*)d_in[2],
                            (const long long*)d_in[3]};
  const float* Wl[3][2];
  const float* bl[3][2];
  {
    int k = 4;
    for (int r = 0; r < 3; ++r)
      for (int l = 0; l < 2; ++l) {
        Wl[r][l] = (const float*)d_in[k++];
        bl[r][l] = (const float*)d_in[k++];
      }
  }
  const float* Wo  = (const float*)d_in[16];
  const float* bo  = (const float*)d_in[17];
  const float* Wih = (const float*)d_in[18];
  const float* Whh = (const float*)d_in[19];
  const float* bih = (const float*)d_in[20];
  const float* bhh = (const float*)d_in[21];

  // workspace layout (bytes); GCN-phase buffers overlap LSTM-phase buffers
  char* ws = (char*)d_ws;
  const size_t SZ_BF  = (size_t)NF * 2;   // 5.12 MB
  const size_t SZ_F   = (size_t)NF * 4;   // 10.24 MB
  const size_t OFF_XB   = 0;              // xb   (later: hsb)
  const size_t OFF_H1B  = OFF_XB + SZ_BF; // h1b  (later: accb)
  const size_t OFF_T    = OFF_H1B + SZ_BF;
  const size_t OFF_ACC  = OFF_T + SZ_F;
  const size_t OFF_AGG  = OFF_ACC + SZ_F; // agg  (later: hs)
  const size_t OFF_G    = OFF_AGG + SZ_F;            // N*512 f32 = 40.96 MB
  const size_t OFF_DEG  = OFF_G + (size_t)N * 512 * 4;
  const size_t OFF_DEGC = OFF_DEG + (size_t)N * 4;
  const size_t OFF_WT   = OFF_DEGC + (size_t)N * 4;  // 128*128 bf16
  const size_t OFF_WIHB = OFF_WT + 128 * 128 * 2;    // 512*128 bf16
  const size_t OFF_BSUM = OFF_WIHB + 512 * 128 * 2;  // 512 f32
  const size_t OFF_WOB  = OFF_BSUM + 512 * 4;        // 64*128 bf16
  const size_t OFF_Y0   = OFF_WOB + 64 * 128 * 2;    // N*64 f32
  const size_t OFF_AGGO = OFF_Y0 + (size_t)N * 64 * 4;

  unsigned short* xb   = (unsigned short*)(ws + OFF_XB);
  unsigned short* h1b  = (unsigned short*)(ws + OFF_H1B);
  float* tbuf          = (float*)(ws + OFF_T);
  float* acc           = (float*)(ws + OFF_ACC);
  float* agg           = (float*)(ws + OFF_AGG);
  float* Gm            = (float*)(ws + OFF_G);
  float* deg           = (float*)(ws + OFF_DEG);
  float* degc          = (float*)(ws + OFF_DEGC);
  unsigned short* Wt   = (unsigned short*)(ws + OFF_WT);
  unsigned short* Wihb = (unsigned short*)(ws + OFF_WIHB);
  float* bsum          = (float*)(ws + OFF_BSUM);
  unsigned short* Wob  = (unsigned short*)(ws + OFF_WOB);
  float* y0            = (float*)(ws + OFF_Y0);
  float* aggo          = (float*)(ws + OFF_AGGO);
  unsigned short* hsb  = xb;   // reuse after GCN phase
  unsigned short* accb = h1b;  // reuse after GCN phase
  float* hs            = agg;  // reuse after GCN phase

  const int EW = 256;
  const int gNF = (NF + EW - 1) / EW;          // 2.56M elementwise
  const int gN  = (N + EW - 1) / EW;
  const dim3 gemmB(256);
  const int SC_BLOCKS = 8192;                  // grid-stride wave-per-edge scatter

  // x -> bf16, acc = 0
  k_f32_to_bf16<<<gNF, EW, 0, stream>>>(x, xb, NF);
  hipMemsetAsync(acc, 0, SZ_F, stream);

  // ---- 3 relations x 2 GCN layers ----
  for (int r = 0; r < 3; ++r) {
    const long long* src = ei[r];
    const long long* tgt = ei[r] + E;
    k_deg_init<<<gN, EW, 0, stream>>>(deg, N);
    k_deg_acc<<<2048, EW, 0, stream>>>(deg, tgt, E);
    k_deg_rsqrt<<<gN, EW, 0, stream>>>(deg, N);  // deg -> dinv in place

    // layer 0: t = x @ W0 ; agg = scatter(norm * t) ; h1 = relu(agg + self + b0)
    k_transpose_bf16<<<64, EW, 0, stream>>>(Wl[r][0], Wt, 128, 128);
    k_gemm_bf16<<<dim3(N / 16, 1), gemmB, 0, stream>>>(xb, Wt, nullptr, tbuf, N, 128);
    hipMemsetAsync(agg, 0, SZ_F, stream);
    k_scatter128<<<SC_BLOCKS, EW, 0, stream>>>(tbuf, src, tgt, deg, agg, E);
    k_finalize_bf16<<<gNF, EW, 0, stream>>>(agg, tbuf, deg, bl[r][0], h1b, NF);

    // layer 1: same, accumulated into acc
    k_transpose_bf16<<<64, EW, 0, stream>>>(Wl[r][1], Wt, 128, 128);
    k_gemm_bf16<<<dim3(N / 16, 1), gemmB, 0, stream>>>(h1b, Wt, nullptr, tbuf, N, 128);
    hipMemsetAsync(agg, 0, SZ_F, stream);
    k_scatter128<<<SC_BLOCKS, EW, 0, stream>>>(tbuf, src, tgt, deg, agg, E);
    k_finalize_acc<<<gNF, EW, 0, stream>>>(agg, tbuf, deg, bl[r][1], acc, NF);
  }

  // ---- LSTM: G = acc @ Wih.T + (bih + bhh), then serial scan ----
  k_f32_to_bf16<<<gNF, EW, 0, stream>>>(acc, accb, NF);
  k_f32_to_bf16<<<(512 * 128 + EW - 1) / EW, EW, 0, stream>>>(Wih, Wihb, 512 * 128);
  k_bias_sum<<<2, EW, 0, stream>>>(bih, bhh, bsum, 512);
  // Wih is [512,128] row-major == exactly the Bt layout for B = Wih.T
  k_gemm_bf16<<<dim3(N / 16, 4), gemmB, 0, stream>>>(accb, Wihb, bsum, Gm, N, 512);
  k_lstm<<<1, 1024, 0, stream>>>(Gm, Whh, hs, N);

  // ---- final GCN over all edges + log_softmax ----
  k_relu_bf16<<<gNF, EW, 0, stream>>>(hs, hsb, NF);
  k_transpose_bf16<<<(128 * 64 + EW - 1) / EW, EW, 0, stream>>>(Wo, Wob, 128, 64);
  k_gemm_bf16<<<dim3(N / 16, 1), gemmB, 0, stream>>>(hsb, Wob, nullptr, y0, N, 64);

  k_deg_init<<<gN, EW, 0, stream>>>(degc, N);
  for (int r = 0; r < 3; ++r)
    k_deg_acc<<<2048, EW, 0, stream>>>(degc, ei[r] + E, E);
  k_deg_rsqrt<<<gN, EW, 0, stream>>>(degc, N);

  hipMemsetAsync(aggo, 0, (size_t)N * 64 * 4, stream);
  for (int r = 0; r < 3; ++r)
    k_scatter64<<<SC_BLOCKS, EW, 0, stream>>>(y0, ei[r], ei[r] + E, degc, aggo, E);

  k_final_logsoftmax<<<(N * 32 + EW - 1) / EW, EW, 0, stream>>>(
      aggo, y0, degc, bo, (float*)d_out, N);
}
